// Seq2SeqDualModelAlignTimestep_40630390620993
// MI455X (gfx1250) — matrix operand order repaired and checked by reading, not compiled, not run on hardware
//
#include <hip/hip_runtime.h>

// ---------------------------------------------------------------------------
// MI455X (gfx1250) Seq2SeqDualModelAlignTimestep forward.
//  - f16-WMMA (fp32 acc) for every GEMM; 32x64 macro-tile per wave (MT2xNT4)
//    so align_prob (256MB) is streamed from HBM exactly once (~11us @ 23.3TB/s).
//  - Persistent wave32 LSTM kernels keep only the recurrent GEMM on the
//    critical path; input projections hoisted into big GEMMs.
//  - enc_out @ Wa runs through a double-buffered TDM (tensor_load_to_lds)
//    staged GEMM with s_wait_tensorcnt pipelining.
// ---------------------------------------------------------------------------

typedef __attribute__((ext_vector_type(16))) _Float16 v16h;
typedef __attribute__((ext_vector_type(8)))  float    v8f;

#define WMMA(acc, a, b) \
  __builtin_amdgcn_wmma_f32_16x16x32_f16(false, (a), false, (b), (short)0, (acc), false, false)

#if defined(__gfx1250__) && __has_builtin(__builtin_amdgcn_tensor_load_to_lds) && \
    __has_builtin(__builtin_amdgcn_s_wait_tensorcnt)
#define HAVE_TDM 1
typedef unsigned int tdm_u32x4 __attribute__((ext_vector_type(4)));
typedef int          tdm_i32x4 __attribute__((ext_vector_type(4)));
typedef int          tdm_i32x8 __attribute__((ext_vector_type(8)));
#endif

namespace cfg {
constexpr int V  = 20000;
constexpr int SE = 400;
constexpr int FR = 60;
constexpr int EIN = SE + FR;   // 460
constexpr int H  = 512;
constexpr int HH = 256;
constexpr int B  = 32;
constexpr int T  = 32;
constexpr int S  = 100;
constexpr int TE = 300;
}

__device__ __forceinline__ float sigf(float x) { return 1.0f / (1.0f + expf(-x)); }

// ---------------------------------------------------------------------------
// Fragment builders (contiguous k = 16*half + s in both A and B), fp32 -> f16 in-register.
// ---------------------------------------------------------------------------

// A 16x32 (MxK): four 16-byte loads per lane (lda % 4 == 0 everywhere).
__device__ __forceinline__ v16h frag_a_full(const float* __restrict__ A, int lda,
                                            int m0, int k0, int lane) {
  const int half = lane >> 4, row = lane & 15;
  const float* p = A + (size_t)(m0 + row) * lda + k0 + half * 16;
  const float4 q0 = *(const float4*)(p);
  const float4 q1 = *(const float4*)(p + 4);
  const float4 q2 = *(const float4*)(p + 8);
  const float4 q3 = *(const float4*)(p + 12);
  v16h a;
  a[0]  = (_Float16)q0.x; a[1]  = (_Float16)q0.y;
  a[2]  = (_Float16)q0.z; a[3]  = (_Float16)q0.w;
  a[4]  = (_Float16)q1.x; a[5]  = (_Float16)q1.y;
  a[6]  = (_Float16)q1.z; a[7]  = (_Float16)q1.w;
  a[8]  = (_Float16)q2.x; a[9]  = (_Float16)q2.y;
  a[10] = (_Float16)q2.z; a[11] = (_Float16)q2.w;
  a[12] = (_Float16)q3.x; a[13] = (_Float16)q3.y;
  a[14] = (_Float16)q3.z; a[15] = (_Float16)q3.w;
  return a;
}

// B 32x16 (KxN): strided per lane, coalesced across lanes. Unpredicated.
__device__ __forceinline__ v16h frag_b_full(const float* __restrict__ Bp, int ldb,
                                            int k0, int n0, int lane) {
  const int half = lane >> 4, col = lane & 15;
  const float* p = Bp + (size_t)(k0 + half * 16) * ldb + n0 + col;
  v16h b;
#pragma unroll
  for (int v = 0; v < 8; ++v) {
    b[2 * v]     = (_Float16)p[0];
    b[2 * v + 1] = (_Float16)p[(size_t)ldb];
    p += 2 * (size_t)ldb;
  }
  return b;
}

// Tail variants: clamped-index (always in-bounds) loads + selects, so the
// loads stay unpredicated/clusterable; zeros merged in with v_cndmask.
__device__ __forceinline__ v16h frag_a_tail(const float* __restrict__ A, int lda,
                                            int m0, int k0, int K, int lane) {
  const int half = lane >> 4, row = lane & 15;
  const float* Ar = A + (size_t)(m0 + row) * lda;
  v16h a;
#pragma unroll
  for (int v = 0; v < 8; ++v) {
    const int kb = k0 + (half << 4) + (v << 1);
    const float x0 = Ar[kb < K ? kb : K - 1];
    const float x1 = Ar[kb + 1 < K ? kb + 1 : K - 1];
    a[2 * v]     = (kb     < K) ? (_Float16)x0 : (_Float16)0.0f;
    a[2 * v + 1] = (kb + 1 < K) ? (_Float16)x1 : (_Float16)0.0f;
  }
  return a;
}

__device__ __forceinline__ v16h frag_b_tail(const float* __restrict__ Bp, int ldb,
                                            int k0, int n0, int K, int lane) {
  const int half = lane >> 4, col = lane & 15;
  v16h b;
#pragma unroll
  for (int v = 0; v < 8; ++v) {
    const int k = k0 + (half << 4) + (v << 1);
    const float x0 = Bp[(size_t)(k     < K ? k     : K - 1) * ldb + n0 + col];
    const float x1 = Bp[(size_t)(k + 1 < K ? k + 1 : K - 1) * ldb + n0 + col];
    b[2 * v]     = (k     < K) ? (_Float16)x0 : (_Float16)0.0f;
    b[2 * v + 1] = (k + 1 < K) ? (_Float16)x1 : (_Float16)0.0f;
  }
  return b;
}

// B fragment out of an LDS-staged [32 x panelW] fp32 tile.
__device__ __forceinline__ v16h frag_b_lds(const float* bt, int panelW, int n0w,
                                           int lane) {
  const int half = lane >> 4, col = lane & 15;
  const float* p = bt + (half * 16) * panelW + n0w + col;
  v16h b;
#pragma unroll
  for (int v = 0; v < 8; ++v) {
    b[2 * v]     = (_Float16)p[0];
    b[2 * v + 1] = (_Float16)p[panelW];
    p += 2 * panelW;
  }
  return b;
}

// ---------------------------------------------------------------------------
// 32x64 macro-tile GEMM core: MT=2 M-tiles x up-to-4 N-tiles per wave.
// nTiles is wave-uniform; call with constant 4 on the fast path so the inner
// loops fully unroll and the 4 B-fragments' loads cluster.
// ---------------------------------------------------------------------------
__device__ __forceinline__ void gemm_macro(const float* __restrict__ A,
                                           const float* __restrict__ Bm,
                                           const float* __restrict__ bias,
                                           float* __restrict__ C,
                                           int K, int lda, int ldb, int ldc,
                                           int m0, int n0, int lane, int nTiles) {
  v8f acc[2][4] = {};
  const int kFull = K & ~31;
  for (int k0 = 0; k0 < kFull; k0 += 32) {
    if (k0 + 32 < K)
      __builtin_prefetch(&Bm[(size_t)(k0 + 32) * ldb + n0 + (lane & 15)], 0, 1);
    const v16h a0 = frag_a_full(A, lda, m0, k0, lane);
    const v16h a1 = frag_a_full(A, lda, m0 + 16, k0, lane);
    for (int nt = 0; nt < nTiles; ++nt) {
      const v16h b = frag_b_full(Bm, ldb, k0, n0 + nt * 16, lane);
      acc[0][nt] = WMMA(acc[0][nt], a0, b);
      acc[1][nt] = WMMA(acc[1][nt], a1, b);
    }
  }
  if (kFull < K) {
    const v16h a0 = frag_a_tail(A, lda, m0, kFull, K, lane);
    const v16h a1 = frag_a_tail(A, lda, m0 + 16, kFull, K, lane);
    for (int nt = 0; nt < nTiles; ++nt) {
      const v16h b = frag_b_tail(Bm, ldb, kFull, n0 + nt * 16, K, lane);
      acc[0][nt] = WMMA(acc[0][nt], a0, b);
      acc[1][nt] = WMMA(acc[1][nt], a1, b);
    }
  }
  const int half = lane >> 4, col = lane & 15;
  for (int nt = 0; nt < nTiles; ++nt) {
    const float bv = bias ? bias[n0 + nt * 16 + col] : 0.0f;
#pragma unroll
    for (int mt = 0; mt < 2; ++mt)
#pragma unroll
      for (int v = 0; v < 8; ++v)
        C[(size_t)(m0 + mt * 16 + v + 8 * half) * ldc + n0 + nt * 16 + col] =
            acc[mt][nt][v] + bv;
  }
}

// Generic batched GEMM: C[b] = A[b] @ B[b] (+ bias). Requires M % 32 == 0.
__global__ void gemm_wmma_kernel(const float* __restrict__ A0,
                                 const float* __restrict__ B0,
                                 const float* __restrict__ bias,
                                 float* __restrict__ C0,
                                 int M, int N, int K, int lda, int ldb, int ldc,
                                 size_t sA, size_t sB, size_t sC) {
  const int tilesN = N >> 4;
  const int groupsN = (tilesN + 3) >> 2;
  const int total = (M >> 5) * groupsN;
  const int wid = blockIdx.x * (blockDim.x >> 5) + (threadIdx.x >> 5);
  if (wid >= total) return;
  const int lane = threadIdx.x & 31;
  const float* A = A0 + (size_t)blockIdx.y * sA;
  const float* Bm = B0 + (size_t)blockIdx.y * sB;
  float* C = C0 + (size_t)blockIdx.y * sC;

  const int m0 = (wid / groupsN) << 5;
  const int ng = wid % groupsN;
  const int n0 = ng << 6;
  const int rem = tilesN - (ng << 2);
  const int nTiles = rem < 4 ? rem : 4;
  if (nTiles == 4) gemm_macro(A, Bm, bias, C, K, lda, ldb, ldc, m0, n0, lane, 4);
  else             gemm_macro(A, Bm, bias, C, K, lda, ldb, ldc, m0, n0, lane, nTiles);
}

// ---------------------------------------------------------------------------
// TDM-staged GEMM (no bias), double-buffered: DMA panel i+1 while computing
// panel i; in-order TDM completion makes s_wait_tensorcnt(1) wait for the
// oldest transfer only. Requires K % 32 == 0, N % 128 == 0.
// ---------------------------------------------------------------------------
#if defined(HAVE_TDM)
__device__ __forceinline__ void tdm_issue_panel(const float* Bm, int ldb, int N,
                                                int K, int k0, int nbase,
                                                const float* ldsdst) {
  const unsigned long long gaddr =
      (unsigned long long)(const void*)(Bm + (size_t)k0 * ldb + nbase);
  const unsigned int lds_lo = (unsigned int)(unsigned long long)(const void*)ldsdst;
  const unsigned int nrem = (unsigned int)(N - nbase);
  const unsigned int krem = (unsigned int)(K - k0);
  tdm_u32x4 g0;
  g0[0] = 1u;                                     // count=1, user mode
  g0[1] = lds_lo;                                 // lds_addr (bytes)
  g0[2] = (unsigned int)gaddr;                    // global_addr[31:0]
  g0[3] = (unsigned int)((gaddr >> 32) & 0x1FFFFFFu) | (2u << 30);  // type=2
  tdm_i32x8 g1;
  g1[0] = (int)(2u << 16);                        // data_size = 4 bytes
  g1[1] = (int)((nrem & 0xFFFFu) << 16);          // tensor_dim0 lo16
  g1[2] = (int)(((nrem >> 16) & 0xFFFFu) | ((krem & 0xFFFFu) << 16));
  g1[3] = (int)(((krem >> 16) & 0xFFFFu) | (128u << 16));  // tile_dim0 = 128
  g1[4] = 32;                                     // tile_dim1 = 32, tile_dim2 = 0
  g1[5] = (int)(unsigned)ldb;                     // tensor_dim0_stride lo32
  g1[6] = 0;
  g1[7] = 0;
  tdm_i32x4 gz = {0, 0, 0, 0};
#if __clang_major__ >= 23
  tdm_i32x8 gz8 = {0, 0, 0, 0, 0, 0, 0, 0};
  __builtin_amdgcn_tensor_load_to_lds(g0, g1, gz, gz, gz8, 0);
#else
  __builtin_amdgcn_tensor_load_to_lds(g0, g1, gz, gz, 0);
#endif
}
#endif

__global__ void gemm_wmma_tdm_kernel(const float* __restrict__ A,
                                     const float* __restrict__ Bm,
                                     float* __restrict__ C,
                                     int M, int N, int K, int lda, int ldb, int ldc) {
  constexpr int PW = 128;  // panel width (N elements per block)
  __shared__ float bt[2][32 * PW];
  const int panelsN = N / PW;
  const int pm = blockIdx.x / panelsN;
  const int pn = blockIdx.x % panelsN;
  const int m0 = pm << 4;
  const int nbase = pn * PW;
  const int wave = threadIdx.x >> 5, lane = threadIdx.x & 31;
  const int n0w = wave << 4;  // 8 waves x 16 columns = 128
  const int nsteps = K / 32;

#if defined(HAVE_TDM)
  if (wave == 0) tdm_issue_panel(Bm, ldb, N, K, 0, nbase, &bt[0][0]);
#endif
  v8f acc = {};
  for (int i = 0; i < nsteps; ++i) {
    const int k0 = i * 32;
#if defined(HAVE_TDM)
    __syncthreads();  // everyone done reading the buffer we are about to fill
    if (wave == 0) {
      if (i + 1 < nsteps) {
        tdm_issue_panel(Bm, ldb, N, K, k0 + 32, nbase, &bt[(i + 1) & 1][0]);
        __builtin_amdgcn_s_wait_tensorcnt(1);  // oldest (panel i) complete
      } else {
        __builtin_amdgcn_s_wait_tensorcnt(0);
      }
    }
    __syncthreads();
    const float* btc = &bt[i & 1][0];
#else
    __syncthreads();
    for (int idx = threadIdx.x; idx < 32 * PW; idx += 256) {
      const int kr = idx / PW, nc = idx - kr * PW;
      bt[0][idx] = Bm[(size_t)(k0 + kr) * ldb + nbase + nc];
    }
    __syncthreads();
    const float* btc = &bt[0][0];
#endif
    const v16h a = frag_a_full(A, lda, m0, k0, lane);
    const v16h b = frag_b_lds(btc, PW, n0w, lane);
    acc = WMMA(acc, a, b);
  }
  const int half = lane >> 4, col = lane & 15;
#pragma unroll
  for (int v = 0; v < 8; ++v)
    C[(size_t)(m0 + v + 8 * half) * ldc + nbase + n0w + col] = acc[v];
}

// ---------------------------------------------------------------------------
// Embedding gathers: enc_in [B*S,460], input_z [B*S,60], x_emb [B*T,400]
// ---------------------------------------------------------------------------
__global__ void embed_kernel(const int* __restrict__ sent, const int* __restrict__ value,
                             const int* __restrict__ field, const int* __restrict__ ppos,
                             const int* __restrict__ pneg,
                             const float* __restrict__ sent_emb, const float* __restrict__ field_emb,
                             const float* __restrict__ ppos_emb, const float* __restrict__ pneg_emb,
                             float* __restrict__ enc_in, float* __restrict__ input_z,
                             float* __restrict__ x_emb) {
  using namespace cfg;
  const int total1 = B * S * EIN;
  const int total  = total1 + B * T * SE;
  for (int i = blockIdx.x * blockDim.x + threadIdx.x; i < total;
       i += gridDim.x * blockDim.x) {
    if (i < total1) {
      const int row = i / EIN, col = i - row * EIN;
      float v;
      if (col < SE) {
        v = sent_emb[(size_t)value[row] * SE + col];
      } else {
        const int c2 = col - SE;
        if (c2 < 50)       v = field_emb[field[row] * 50 + c2];
        else if (c2 < 55)  v = ppos_emb[ppos[row] * 5 + (c2 - 50)];
        else               v = pneg_emb[pneg[row] * 5 + (c2 - 55)];
        input_z[(size_t)row * FR + c2] = v;
      }
      enc_in[(size_t)row * EIN + col] = v;
    } else {
      const int j = i - total1;
      const int row = j / SE, col = j - row * SE;
      x_emb[(size_t)row * SE + col] = sent_emb[(size_t)sent[row] * SE + col];
    }
  }
}

// ---------------------------------------------------------------------------
// Bidirectional encoder LSTM: persistent block per direction, 16 waves.
// ---------------------------------------------------------------------------
__global__ void encoder_lstm_kernel(const float* __restrict__ XWi_f, const float* __restrict__ XWi_b,
                                    const float* __restrict__ Wh_f, const float* __restrict__ Wh_b,
                                    float* __restrict__ enc_out,
                                    float* h_f, float* c_f, float* h_b, float* c_b,
                                    float* z_f, float* z_b) {
  using namespace cfg;
  const int dir = blockIdx.x;
  const float* XWi = dir ? XWi_b : XWi_f;
  const float* Wh  = dir ? Wh_b  : Wh_f;
  float* h = dir ? h_b : h_f;
  float* c = dir ? c_b : c_f;
  float* z = dir ? z_b : z_f;
  const int tid = threadIdx.x;
  const int wave = tid >> 5, lane = tid & 31;

  for (int i = tid; i < B * HH; i += 512) { h[i] = 0.0f; c[i] = 0.0f; }
  __syncthreads();

  for (int step = 0; step < S; ++step) {
    const int s = dir ? (S - 1 - step) : step;
    for (int ti = wave; ti < 128; ti += 16) {   // 2 x 64 tiles
      const int m0 = (ti >> 6) << 4;
      const int n0 = (ti & 63) << 4;
      v8f acc = {};
#pragma unroll
      for (int k0 = 0; k0 < HH; k0 += 32) {
        const v16h a = frag_a_full(h, HH, m0, k0, lane);
        const v16h b = frag_b_full(Wh, 4 * HH, k0, n0, lane);
        acc = WMMA(acc, a, b);
      }
      const int half = lane >> 4, col = lane & 15;
#pragma unroll
      for (int v = 0; v < 8; ++v) {
        const int bb = m0 + v + 8 * half;
        const int n = n0 + col;
        z[(size_t)bb * 1024 + n] = acc[v] + XWi[(size_t)(bb * S + s) * 1024 + n];
      }
    }
    __syncthreads();
    for (int i = tid; i < B * HH; i += 512) {   // gates, torch order i,f,g,o
      const int bb = i >> 8, j = i & 255;
      const float* zr = z + (size_t)bb * 1024;
      const float zi = zr[j], zf = zr[256 + j], zg = zr[512 + j], zo = zr[768 + j];
      const float cv = sigf(zf) * c[i] + sigf(zi) * tanhf(zg);
      const float hv = sigf(zo) * tanhf(cv);
      c[i] = cv; h[i] = hv;
      enc_out[(size_t)(bb * S + s) * H + dir * HH + j] = hv;
    }
    __syncthreads();
  }
}

// ---------------------------------------------------------------------------
// Dual-attention decoder: persistent single block, 32 waves.
// ---------------------------------------------------------------------------
__global__ void decoder_kernel(const float* __restrict__ XWi_d, const float* __restrict__ Wh_d,
                               const float* __restrict__ Wo, const float* __restrict__ Wl,
                               const float* __restrict__ bl,
                               const float* __restrict__ enc_proj, const float* __restrict__ z_proj,
                               const float* __restrict__ enc_out, const float* __restrict__ x_emb,
                               const float* __restrict__ h_f, const float* __restrict__ h_b,
                               const float* __restrict__ c_f, const float* __restrict__ c_b,
                               float* dh, float* dc, float* dz, float* cat,
                               float* __restrict__ outs, float* __restrict__ attn_w,
                               float* __restrict__ lam_w,
                               float* __restrict__ out_hT, float* __restrict__ out_cT) {
  using namespace cfg;
  __shared__ float sA[B * S];
  __shared__ float sBt[B * S];
  __shared__ float sG[B * S];
  const int tid = threadIdx.x;
  const int wave = tid >> 5, lane = tid & 31;

  for (int i = tid; i < B * H; i += 1024) {
    const int b = i >> 9, j = i & 511;
    dh[i] = (j < HH) ? h_f[b * HH + j] : h_b[b * HH + (j - HH)];
    dc[i] = (j < HH) ? c_f[b * HH + j] : c_b[b * HH + (j - HH)];
  }
  __syncthreads();

  for (int t = 0; t < T; ++t) {
    for (int ti = wave; ti < 256; ti += 32) {   // 2 x 128 tiles
      const int m0 = (ti >> 7) << 4;
      const int n0 = (ti & 127) << 4;
      v8f acc = {};
      for (int k0 = 0; k0 < H; k0 += 32) {
        const v16h a = frag_a_full(dh, H, m0, k0, lane);
        const v16h b = frag_b_full(Wh_d, 4 * H, k0, n0, lane);
        acc = WMMA(acc, a, b);
      }
      const int half = lane >> 4, col = lane & 15;
#pragma unroll
      for (int v = 0; v < 8; ++v) {
        const int bb = m0 + v + 8 * half;
        const int n = n0 + col;
        dz[(size_t)bb * 2048 + n] = acc[v] + XWi_d[(size_t)(bb * T + t) * 2048 + n];
      }
    }
    __syncthreads();
    for (int i = tid; i < B * H; i += 1024) {   // gates
      const int b = i >> 9, j = i & 511;
      const float* zr = dz + (size_t)b * 2048;
      const float zi = zr[j], zf = zr[512 + j], zg = zr[1024 + j], zo = zr[1536 + j];
      const float cv = sigf(zf) * dc[i] + sigf(zi) * tanhf(zg);
      const float hv = sigf(zo) * tanhf(cv);
      dc[i] = cv; dh[i] = hv;
      cat[(size_t)b * 1024 + j] = hv;
    }
    __syncthreads();
    for (int i = tid; i < B * S; i += 1024) {   // attention logits (float4 dots)
      const int b = i / S, s = i - b * S;
      const float4* ep = (const float4*)(enc_proj + (size_t)(b * S + s) * H);
      const float4* zp = (const float4*)(z_proj + (size_t)(b * S + s) * H);
      const float4* hb = (const float4*)(dh + (size_t)b * H);
      float da = 0.0f, db = 0.0f;
      for (int k = 0; k < H / 4; ++k) {
        const float4 h4 = hb[k];
        const float4 e4 = ep[k];
        const float4 z4 = zp[k];
        da += e4.x * h4.x + e4.y * h4.y + e4.z * h4.z + e4.w * h4.w;
        db += z4.x * h4.x + z4.y * h4.y + z4.z * h4.z + z4.w * h4.w;
      }
      sA[i] = da; sBt[i] = db;
    }
    __syncthreads();
    if (tid < B) {   // dual softmax + renormalized product
      float ma = -3.4e38f, mb = -3.4e38f;
      for (int s = 0; s < S; ++s) {
        ma = fmaxf(ma, sA[tid * S + s]);
        mb = fmaxf(mb, sBt[tid * S + s]);
      }
      float sa = 0.0f, sb = 0.0f;
      for (int s = 0; s < S; ++s) {
        sa += expf(sA[tid * S + s] - ma);
        sb += expf(sBt[tid * S + s] - mb);
      }
      float gs = 0.0f;
      for (int s = 0; s < S; ++s) {
        const float g = (expf(sA[tid * S + s] - ma) / sa) *
                        (expf(sBt[tid * S + s] - mb) / sb);
        sG[tid * S + s] = g;
        gs += g;
      }
      const float inv = 1.0f / (gs + 1e-9f);
      for (int s = 0; s < S; ++s) {
        const float g = sG[tid * S + s] * inv;
        sG[tid * S + s] = g;
        attn_w[(size_t)(tid * T + t) * S + s] = g;
      }
    }
    __syncthreads();
    for (int i = tid; i < B * H; i += 1024) {   // context
      const int b = i >> 9, j = i & 511;
      float cx = 0.0f;
      for (int s = 0; s < S; ++s)
        cx += sG[b * S + s] * enc_out[(size_t)(b * S + s) * H + j];
      cat[(size_t)b * 1024 + 512 + j] = cx;
    }
    __syncthreads();
    for (int ti = wave; ti < 64; ti += 32) {    // out = tanh([h;ctx] @ Wo)
      const int m0 = (ti >> 5) << 4;
      const int n0 = (ti & 31) << 4;
      v8f acc = {};
      for (int k0 = 0; k0 < 1024; k0 += 32) {
        const v16h a = frag_a_full(cat, 1024, m0, k0, lane);
        const v16h b = frag_b_full(Wo, H, k0, n0, lane);
        acc = WMMA(acc, a, b);
      }
      const int half = lane >> 4, col = lane & 15;
#pragma unroll
      for (int v = 0; v < 8; ++v) {
        const int bb = m0 + v + 8 * half;
        outs[(size_t)(bb * T + t) * H + n0 + col] = tanhf(acc[v]);
      }
    }
    if (tid < B) {   // lambda gate
      const float* hb = dh + (size_t)tid * H;
      const float* cx = cat + (size_t)tid * 1024 + 512;
      const float* xt = x_emb + (size_t)(tid * T + t) * SE;
      float a = bl[0];
      for (int k = 0; k < H; ++k)  a += hb[k] * Wl[k];
      for (int k = 0; k < H; ++k)  a += cx[k] * Wl[512 + k];
      for (int k = 0; k < SE; ++k) a += xt[k] * Wl[1024 + k];
      lam_w[tid * T + t] = sigf(a);
    }
    if (t == T - 1) {
      for (int i = tid; i < B * H; i += 1024) { out_hT[i] = dh[i]; out_cT[i] = dc[i]; }
    }
    __syncthreads();
  }
}

// ---------------------------------------------------------------------------
// Fused: p_mod = log_softmax(logits); out = lam*p_lex + (1-lam)*p_mod
// All passes vectorized float4 (V % 4 == 0, rows 16B-aligned).
// ---------------------------------------------------------------------------
__global__ void mix_logsoftmax_kernel(const float* __restrict__ logits,
                                      const float* __restrict__ p_lex,
                                      const float* __restrict__ lam,
                                      float* __restrict__ out) {
  using namespace cfg;
  const int r = blockIdx.x, tid = threadIdx.x;
  __shared__ float red[256];
  const float4* lr = (const float4*)(logits + (size_t)r * V);
  const int V4 = V / 4;
  float m = -3.4e38f;
  for (int v = tid; v < V4; v += 256) {
    const float4 x = lr[v];
    m = fmaxf(m, fmaxf(fmaxf(x.x, x.y), fmaxf(x.z, x.w)));
  }
  red[tid] = m; __syncthreads();
  for (int st = 128; st > 0; st >>= 1) {
    if (tid < st) red[tid] = fmaxf(red[tid], red[tid + st]);
    __syncthreads();
  }
  const float rowmax = red[0]; __syncthreads();
  float s = 0.0f;
  for (int v = tid; v < V4; v += 256) {
    const float4 x = lr[v];
    s += expf(x.x - rowmax) + expf(x.y - rowmax) + expf(x.z - rowmax) + expf(x.w - rowmax);
  }
  red[tid] = s; __syncthreads();
  for (int st = 128; st > 0; st >>= 1) {
    if (tid < st) red[tid] += red[tid + st];
    __syncthreads();
  }
  const float lse = rowmax + logf(red[0]);
  const float L = lam[r], Lc = 1.0f - L;
  const float4* px = (const float4*)(p_lex + (size_t)r * V);
  float4* o = (float4*)(out + (size_t)r * V);
  for (int v = tid; v < V4; v += 256) {
    const float4 x = lr[v];
    const float4 p = px[v];
    float4 y;
    y.x = L * p.x + Lc * (x.x - lse);
    y.y = L * p.y + Lc * (x.y - lse);
    y.z = L * p.z + Lc * (x.z - lse);
    y.w = L * p.w + Lc * (x.w - lse);
    o[v] = y;
  }
}

// ---------------------------------------------------------------------------
// In-place row-wise log_softmax over p_bias + per-row argmax (float4)
// ---------------------------------------------------------------------------
__global__ void logsoftmax_argmax_kernel(float* __restrict__ pb, int* __restrict__ idx_out) {
  using namespace cfg;
  const int r = blockIdx.x, tid = threadIdx.x;
  __shared__ float rv[256];
  __shared__ int ri[256];
  float4* x4 = (float4*)(pb + (size_t)r * V);
  const int V4 = V / 4;
  float m = -3.4e38f; int mi = 0;
  for (int v = tid; v < V4; v += 256) {
    const float4 x = x4[v];
    if (x.x > m) { m = x.x; mi = 4 * v; }
    if (x.y > m) { m = x.y; mi = 4 * v + 1; }
    if (x.z > m) { m = x.z; mi = 4 * v + 2; }
    if (x.w > m) { m = x.w; mi = 4 * v + 3; }
  }
  rv[tid] = m; ri[tid] = mi; __syncthreads();
  for (int st = 128; st > 0; st >>= 1) {
    if (tid < st) {
      if (rv[tid + st] > rv[tid] ||
          (rv[tid + st] == rv[tid] && ri[tid + st] < ri[tid])) {
        rv[tid] = rv[tid + st]; ri[tid] = ri[tid + st];
      }
    }
    __syncthreads();
  }
  const float rowmax = rv[0];
  if (tid == 0) idx_out[r] = ri[0];
  __syncthreads();
  float s = 0.0f;
  for (int v = tid; v < V4; v += 256) {
    const float4 x = x4[v];
    s += expf(x.x - rowmax) + expf(x.y - rowmax) + expf(x.z - rowmax) + expf(x.w - rowmax);
  }
  rv[tid] = s; __syncthreads();
  for (int st = 128; st > 0; st >>= 1) {
    if (tid < st) rv[tid] += rv[tid + st];
    __syncthreads();
  }
  const float lse = rowmax + logf(rv[0]);
  for (int v = tid; v < V4; v += 256) {
    float4 x = x4[v];
    x.x -= lse; x.y -= lse; x.z -= lse; x.w -= lse;
    x4[v] = x;
  }
}

// ---------------------------------------------------------------------------
// attn argmax -> value -> target_emb gather; decoder argmax -> target_emb
// ---------------------------------------------------------------------------
__global__ void pred_kernel(const float* __restrict__ attn_w, const int* __restrict__ value,
                            const int* __restrict__ idx_dec, const float* __restrict__ target_emb,
                            float* __restrict__ out_attn, float* __restrict__ out_dec) {
  using namespace cfg;
  const int r = blockIdx.x, tid = threadIdx.x;
  __shared__ int sw[2];
  if (tid == 0) {
    const float* a = attn_w + (size_t)r * S;
    float m = a[0]; int mi = 0;
    for (int s = 1; s < S; ++s)
      if (a[s] > m) { m = a[s]; mi = s; }
    sw[0] = value[(r / T) * S + mi];
    sw[1] = idx_dec[r];
  }
  __syncthreads();
  for (int j = tid; j < TE; j += 128) {
    out_attn[(size_t)r * TE + j] = target_emb[(size_t)sw[0] * TE + j];
    out_dec[(size_t)r * TE + j]  = target_emb[(size_t)sw[1] * TE + j];
  }
}

// ---------------------------------------------------------------------------
// Host orchestration
// ---------------------------------------------------------------------------
static inline void launch_gemm(const float* A, const float* Bm, const float* bias,
                               float* C, int M, int N, int K, int lda, int ldb,
                               int ldc, size_t sA, size_t sB, size_t sC,
                               int batch, hipStream_t stream) {
  const int groupsN = (N / 16 + 3) / 4;
  const int waves = (M / 32) * groupsN;
  const int blocks = (waves + 7) / 8;  // 8 waves (256 threads) per block
  gemm_wmma_kernel<<<dim3(blocks, batch), dim3(256), 0, stream>>>(
      A, Bm, bias, C, M, N, K, lda, ldb, ldc, sA, sB, sC);
}

extern "C" void kernel_launch(void* const* d_in, const int* in_sizes, int n_in,
                              void* d_out, int out_size, void* d_ws, size_t ws_size,
                              hipStream_t stream) {
  using namespace cfg;
  const int*   sent       = (const int*)d_in[0];
  const int*   value      = (const int*)d_in[1];
  const int*   field      = (const int*)d_in[2];
  const int*   ppos       = (const int*)d_in[3];
  const int*   pneg       = (const int*)d_in[4];
  const float* align_prob = (const float*)d_in[7];
  const float* sent_emb   = (const float*)d_in[8];
  const float* field_emb  = (const float*)d_in[9];
  const float* ppos_emb   = (const float*)d_in[10];
  const float* pneg_emb   = (const float*)d_in[11];
  const float* target_emb = (const float*)d_in[12];
  const float* Wi_f = (const float*)d_in[13];
  const float* Wh_f = (const float*)d_in[14];
  const float* b_f  = (const float*)d_in[15];
  const float* Wi_b = (const float*)d_in[16];
  const float* Wh_b = (const float*)d_in[17];
  const float* b_b  = (const float*)d_in[18];
  const float* Wi_d = (const float*)d_in[19];
  const float* Wh_d = (const float*)d_in[20];
  const float* b_d  = (const float*)d_in[21];
  const float* Wa   = (const float*)d_in[22];
  const float* Wf   = (const float*)d_in[23];
  const float* Wo   = (const float*)d_in[24];
  const float* Wl   = (const float*)d_in[25];
  const float* bl   = (const float*)d_in[26];
  const float* Wout = (const float*)d_in[27];
  const float* bout = (const float*)d_in[28];
  (void)in_sizes; (void)n_in; (void)out_size; (void)ws_size;

  // ---- workspace layout (fp32 elements, all offsets 16B-aligned) ----
  float* ws = (float*)d_ws;
  size_t off = 0;
  auto F = [&](size_t n) { float* p = ws + off; off += n; return p; };
  float* enc_in   = F((size_t)B * S * EIN);
  float* input_z  = F((size_t)B * S * FR);
  float* x_emb    = F((size_t)B * T * SE);
  float* XWi_f    = F((size_t)B * S * 4 * HH);
  float* XWi_b    = F((size_t)B * S * 4 * HH);
  float* XWi_d    = F((size_t)B * T * 4 * H);
  float* enc_out  = F((size_t)B * S * H);
  float* z_proj   = F((size_t)B * S * H);
  float* enc_proj = F((size_t)B * S * H);
  float* h_f = F(B * HH); float* c_f = F(B * HH);
  float* h_b = F(B * HH); float* c_b = F(B * HH);
  float* z_f = F(B * 4 * HH); float* z_b = F(B * 4 * HH);
  float* dh  = F(B * H); float* dc = F(B * H);
  float* dz  = F(B * 4 * H); float* cat = F(B * 2 * H);
  float* outs   = F((size_t)B * T * H);
  float* attn_w = F((size_t)B * T * S);
  float* lam_w  = F((size_t)B * T);
  float* logits = F((size_t)B * T * V);
  float* p_lex  = F((size_t)B * T * V);
  int*   idx_dec = (int*)(ws + off); off += B * T;

  // ---- output layout ----
  float* out     = (float*)d_out;
  float* out_pb  = out;
  float* out_hT  = out + (size_t)B * T * V;
  float* out_cT  = out_hT + (size_t)B * H;
  float* out_ap  = out_cT + (size_t)B * H;
  float* out_dp  = out_ap + (size_t)B * T * TE;

  // 1) embedding gathers
  {
    const int n = B * S * EIN + B * T * SE;
    embed_kernel<<<(n + 255) / 256, 256, 0, stream>>>(
        sent, value, field, ppos, pneg, sent_emb, field_emb, ppos_emb, pneg_emb,
        enc_in, input_z, x_emb);
  }
  // 2) hoisted input projections (biases folded in)
  launch_gemm(enc_in, Wi_f, b_f, XWi_f, B * S, 4 * HH, EIN, EIN, 4 * HH, 4 * HH, 0, 0, 0, 1, stream);
  launch_gemm(enc_in, Wi_b, b_b, XWi_b, B * S, 4 * HH, EIN, EIN, 4 * HH, 4 * HH, 0, 0, 0, 1, stream);
  launch_gemm(x_emb,  Wi_d, b_d, XWi_d, B * T, 4 * H,  SE,  SE,  4 * H,  4 * H,  0, 0, 0, 1, stream);
  // 3) field-level attention keys (K=60, clamped tail)
  launch_gemm(input_z, Wf, nullptr, z_proj, B * S, H, FR, FR, H, H, 0, 0, 0, 1, stream);
  // 4) bidirectional encoder LSTM
  encoder_lstm_kernel<<<dim3(2), dim3(512), 0, stream>>>(
      XWi_f, XWi_b, Wh_f, Wh_b, enc_out, h_f, c_f, h_b, c_b, z_f, z_b);
  // 5) word-level attention keys via double-buffered TDM GEMM (3200x512x512)
  {
    const int blocks = (B * S / 16) * (H / 128);
    gemm_wmma_tdm_kernel<<<blocks, 256, 0, stream>>>(
        enc_out, Wa, enc_proj, B * S, H, H, H, H, H);
  }
  // 6) dual-attention decoder
  decoder_kernel<<<dim3(1), dim3(1024), 0, stream>>>(
      XWi_d, Wh_d, Wo, Wl, bl, enc_proj, z_proj, enc_out, x_emb,
      h_f, h_b, c_f, c_b, dh, dc, dz, cat, outs, attn_w, lam_w, out_hT, out_cT);
  // 7) vocab projection: [1024,512] @ [512,20000] + bout
  launch_gemm(outs, Wout, bout, logits, B * T, V, H, H, V, V, 0, 0, 0, 1, stream);
  // 8) p_lex = attn @ align_prob, batched over B; M=32 -> one macro-tile row,
  //    so the 256MB align_prob tensor is streamed from HBM exactly once.
  launch_gemm(attn_w, align_prob, nullptr, p_lex, T, V, S, S, V, V,
              (size_t)T * S, (size_t)S * V, (size_t)T * V, B, stream);
  // 9) p_mod log_softmax fused with lambda mix -> d_out p_bias
  mix_logsoftmax_kernel<<<B * T, 256, 0, stream>>>(logits, p_lex, lam_w, out_pb);
  // 10) final log_softmax in-place + argmax indices
  logsoftmax_argmax_kernel<<<B * T, 256, 0, stream>>>(out_pb, idx_dec);
  // 11) prediction embedding gathers
  pred_kernel<<<B * T, 128, 0, stream>>>(attn_w, value, idx_dec, target_emb,
                                         out_ap, out_dp);
}
